// CAT_8675833938053
// MI455X (gfx1250) — compile-verified, matrix operand order checked by reading, not run on hardware
//
#include <hip/hip_runtime.h>
#include <hip/hip_bf16.h>
#include <math.h>

// ---------------------------------------------------------------------------
// Types for CDNA5 WMMA (wave32): v_wmma_f32_16x16x32_bf16
// ---------------------------------------------------------------------------
typedef __attribute__((ext_vector_type(16))) __bf16 bf16x16;
typedef __attribute__((ext_vector_type(8)))  __bf16 bf16x8;
typedef __attribute__((ext_vector_type(8)))  float  floatx8;

#define SELU_LAMBDA 1.0507009873554804934193349852946f
#define SELU_ALPHA  1.6732632423543772848170429916717f

__device__ __forceinline__ float selu_f(float x) {
  return SELU_LAMBDA * (x > 0.0f ? x : SELU_ALPHA * (__expf(x) - 1.0f));
}

// ---------------------------------------------------------------------------
// GEMM: C[M,Ncols] = A[M,256] @ B[256,Ncols], with
//   A  : bf16 row-major  [M, 256]
//   Bt : bf16 TRANSPOSED [Ncols, 256]   (Bt[n*256+k] = B[k,n])
//   C  : fp32 row-major  [M, Ncols]
// Per ISA 7.12.2 the 16-bit fragment of lane l (half=l/16) is two contiguous
// K-chunks [8*half, +8) and [16+8*half, +8) -> two b128 loads per fragment.
// One wave computes NT 16x16 tiles of one 16-row m-tile; NT is compile-time so
// there is no divergent guard (EXEC stays all-ones around every WMMA).
// ---------------------------------------------------------------------------
template <int NT>
__global__ void __launch_bounds__(128)
gemm_wmma_bf16t(const __bf16* __restrict__ A,
                const __bf16* __restrict__ Bt,
                float* __restrict__ C,
                int M, int Ncols) {
  constexpr int KD = 256;
  const int lane = threadIdx.x & 31;
  const int wave = threadIdx.x >> 5;
  const int wavesPerBlock = blockDim.x >> 5;
  const int m0 = blockIdx.x * 16;
  const int nbase = ((blockIdx.y * wavesPerBlock) + wave) * (16 * NT);
  const int half = lane >> 4;
  const int mr = lane & 15;                 // A row / B,C col within tile

  // hoisted fragment base pointers: chunk0 at +k0, chunk1 at +k0+16
  const __bf16* ap = A + (size_t)(m0 + mr) * KD + 8 * half;
  const __bf16* bp[NT];
#pragma unroll
  for (int t = 0; t < NT; ++t)
    bp[t] = Bt + (size_t)(nbase + 16 * t + mr) * KD + 8 * half;

  floatx8 acc[NT];
#pragma unroll
  for (int t = 0; t < NT; ++t) acc[t] = floatx8{0.f,0.f,0.f,0.f,0.f,0.f,0.f,0.f};

  union Frag { bf16x16 v; bf16x8 h[2]; };

#pragma unroll 2
  for (int k0 = 0; k0 < KD; k0 += 32) {
    Frag af;
    af.h[0] = *(const bf16x8*)(ap + k0);
    af.h[1] = *(const bf16x8*)(ap + k0 + 16);
#pragma unroll
    for (int t = 0; t < NT; ++t) {
      Frag bf;
      bf.h[0] = *(const bf16x8*)(bp[t] + k0);
      bf.h[1] = *(const bf16x8*)(bp[t] + k0 + 16);
      acc[t] = __builtin_amdgcn_wmma_f32_16x16x32_bf16(
          false, af.v, false, bf.v, (short)0, acc[t], false, false);
    }
  }

#pragma unroll
  for (int t = 0; t < NT; ++t) {
#pragma unroll
    for (int r = 0; r < 8; ++r) {
      int m = m0 + r + 8 * half;
      if (m < M) C[(size_t)m * Ncols + nbase + 16 * t + mr] = acc[t][r];
    }
  }
}

// ---------------------------------------------------------------------------
// fp32 -> bf16 conversion (flat) and transpose+convert for weight matrices
// ---------------------------------------------------------------------------
__global__ void cvt_bf16(const float* __restrict__ src, __bf16* __restrict__ dst,
                         long long n) {
  long long i = (long long)blockIdx.x * blockDim.x + threadIdx.x;
  if (i < n) dst[i] = (__bf16)src[i];
}

// Bt[n*Kd + k] = (bf16) B[k*Ncols + n]
__global__ void transpose_cvt_bf16(const float* __restrict__ B, __bf16* __restrict__ Bt,
                                   int Kd, int Ncols) {
  int i = blockIdx.x * blockDim.x + threadIdx.x;
  if (i >= Kd * Ncols) return;
  int k = i / Ncols, n = i % Ncols;
  Bt[(size_t)n * Kd + k] = (__bf16)B[i];
}

// ---------------------------------------------------------------------------
// CSR build: count edges per src (+ degrees per dst), scan, fill
// ---------------------------------------------------------------------------
__global__ void edge_count(const int* __restrict__ src, const int* __restrict__ dst,
                           int* __restrict__ counts, float* __restrict__ deg, int E) {
  int e = blockIdx.x * blockDim.x + threadIdx.x;
  if (e >= E) return;
  atomicAdd(&counts[src[e]], 1);
  atomicAdd(&deg[dst[e]], 1.0f);
}

__global__ void exclusive_scan(const int* __restrict__ counts, int* __restrict__ ptr, int Nn) {
  __shared__ int sh[1024];
  __shared__ int carry;
  if (threadIdx.x == 0) carry = 0;
  __syncthreads();
  for (int base = 0; base < Nn; base += 1024) {
    int i = base + threadIdx.x;
    int v = (i < Nn) ? counts[i] : 0;
    sh[threadIdx.x] = v;
    __syncthreads();
    for (int off = 1; off < 1024; off <<= 1) {
      int t = (threadIdx.x >= off) ? sh[threadIdx.x - off] : 0;
      __syncthreads();
      sh[threadIdx.x] += t;
      __syncthreads();
    }
    if (i < Nn) ptr[i] = carry + sh[threadIdx.x] - v;   // exclusive
    int tot = sh[1023];
    __syncthreads();
    if (threadIdx.x == 0) carry += tot;
    __syncthreads();
  }
  if (threadIdx.x == 0) ptr[Nn] = carry;
}

__global__ void edge_fill(const int* __restrict__ src, const int* __restrict__ dst,
                          const float* __restrict__ w, const int* __restrict__ ptr,
                          int* __restrict__ fill, int* __restrict__ sdst,
                          float* __restrict__ sw, int E) {
  int e = blockIdx.x * blockDim.x + threadIdx.x;
  if (e >= E) return;
  int s = src[e];
  int pos = ptr[s] + atomicAdd(&fill[s], 1);
  sdst[pos] = dst[e];
  sw[pos]   = w[e];
}

// ---------------------------------------------------------------------------
// Wide SpMM: Ybf[n,f] = (bf16) selu( sum_e w[e]*X[dst[e],f] + bias[f] )
// one block per node, one thread per feature (Fd == blockDim.x)
// ---------------------------------------------------------------------------
__global__ void spmm_bias_selu_bf16(const float* __restrict__ X, const int* __restrict__ ptr,
                                    const int* __restrict__ sdst, const float* __restrict__ sw,
                                    const float* __restrict__ bias, __bf16* __restrict__ Y,
                                    int Fd) {
  int node = blockIdx.x;
  int f = threadIdx.x;
  float a = 0.f;
  int beg = ptr[node], end = ptr[node + 1];
  for (int e = beg; e < end; ++e) {
    if (e + 1 < end)
      __builtin_prefetch(&X[(long long)sdst[e + 1] * Fd + f], 0, 1);  // global_prefetch_b8
    a += sw[e] * X[(long long)sdst[e] * Fd + f];
  }
  a += bias[f];
  Y[(long long)node * Fd + f] = (__bf16)selu_f(a);
}

// ---------------------------------------------------------------------------
// Narrow (K=16) SpMM, one thread/node.
// mode 1: weighted, then +bias, selu, softmax (assignment head)
// mode 0: binary adjacency (A @ S), raw output
// ---------------------------------------------------------------------------
__global__ void spmm16_post(const float* __restrict__ X, const int* __restrict__ ptr,
                            const int* __restrict__ sdst, const float* __restrict__ sw,
                            const float* __restrict__ bias, float* __restrict__ Y,
                            int Nn, int mode) {
  int node = blockIdx.x * blockDim.x + threadIdx.x;
  if (node >= Nn) return;
  float v[16];
#pragma unroll
  for (int k = 0; k < 16; ++k) v[k] = 0.f;
  int beg = ptr[node], end = ptr[node + 1];
  for (int e = beg; e < end; ++e) {
    const float* row = &X[(long long)sdst[e] * 16];
    float wv = (mode == 1) ? sw[e] : 1.0f;
#pragma unroll
    for (int k = 0; k < 16; ++k) v[k] += wv * row[k];
  }
  if (mode == 1) {
    float mx = -3.0e38f;
#pragma unroll
    for (int k = 0; k < 16; ++k) { v[k] = selu_f(v[k] + bias[k]); mx = fmaxf(mx, v[k]); }
    float s = 0.f;
#pragma unroll
    for (int k = 0; k < 16; ++k) { v[k] = __expf(v[k] - mx); s += v[k]; }
    float inv = 1.0f / s;
#pragma unroll
    for (int k = 0; k < 16; ++k) v[k] *= inv;
  }
#pragma unroll
  for (int k = 0; k < 16; ++k) Y[(long long)node * 16 + k] = v[k];
}

// ---------------------------------------------------------------------------
// Per-node statistics -> acc[34]: [0]=trace(AS^T S), [1]=sum cosine(hat,S),
// [2..17]=normalizer_left (S^T deg), [18..33]=cluster_sizes
// ---------------------------------------------------------------------------
__global__ void reduce_stats(const float* __restrict__ S, const float* __restrict__ ASm,
                             const float* __restrict__ Hh, const float* __restrict__ deg,
                             float* __restrict__ acc, int Nn) {
  int node = blockIdx.x * blockDim.x + threadIdx.x;
  if (node >= Nn) return;
  const float* s = &S[(long long)node * 16];
  const float* a = &ASm[(long long)node * 16];
  const float* h = &Hh[(long long)node * 16];
  float d = deg[node];
  float tr = 0.f, ns = 0.f, nh = 0.f, dp = 0.f;
#pragma unroll
  for (int k = 0; k < 16; ++k) {
    tr += a[k] * s[k];
    ns += s[k] * s[k];
    nh += h[k] * h[k];
    dp += h[k] * s[k];
    atomicAdd(&acc[2 + k],  s[k] * d);
    atomicAdd(&acc[18 + k], s[k]);
  }
  float cn = dp / (fmaxf(sqrtf(ns), 1e-12f) * fmaxf(sqrtf(nh), 1e-12f));
  atomicAdd(&acc[0], tr);
  atomicAdd(&acc[1], cn);
}

__global__ void finalize_loss(const float* __restrict__ acc, float* __restrict__ out,
                              float Ef, float Nf, float Kf) {
  float tr = acc[0], cons = acc[1];
  float nl2 = 0.f, cs2 = 0.f;
  for (int k = 0; k < 16; ++k) {
    nl2 += acc[2 + k] * acc[2 + k];
    cs2 += acc[18 + k] * acc[18 + k];
  }
  float norm_tr  = nl2 / (2.0f * Ef);
  float spectral = -(tr - norm_tr) / (2.0f * Ef);
  float cluster  = sqrtf(cs2) / Nf * sqrtf(Kf) - 1.0f;
  float con      = 2.0f - 2.0f * (cons / Nf);
  out[0] = spectral + cluster + con;
}

// ---------------------------------------------------------------------------
// Host: orchestrate on `stream` only (graph-capture safe)
// ---------------------------------------------------------------------------
extern "C" void kernel_launch(void* const* d_in, const int* in_sizes, int n_in,
                              void* d_out, int out_size, void* d_ws, size_t ws_size,
                              hipStream_t stream) {
  const int*   esrc = (const int*)d_in[0];
  const int*   edst = (const int*)d_in[1];
  const float* ew   = (const float*)d_in[2];
  const float* feat = (const float*)d_in[3];
  const float* aug  = (const float*)d_in[4];
  const float* W1   = (const float*)d_in[5];
  const float* b1   = (const float*)d_in[6];
  const float* Wst  = (const float*)d_in[7];
  const float* bs   = (const float*)d_in[8];
  const float* Wt   = (const float*)d_in[9];
  const float* bt   = (const float*)d_in[10];

  const int E = in_sizes[0];
  const int F = 256, H = 256, K = 16;
  const int N = in_sizes[3] / F;

  // workspace carve-up (256B aligned)
  char* wp = (char*)d_ws;
  auto alloc = [&](size_t bytes) -> char* {
    char* p = wp; wp += (bytes + 255) & ~(size_t)255; return p;
  };
  float*  XW     = (float*) alloc((size_t)N * H * 4);    // X @ W1 (fp32)
  __bf16* XB     = (__bf16*)alloc((size_t)N * F * 2);    // bf16 features / aug
  __bf16* GB     = (__bf16*)alloc((size_t)N * H * 2);    // bf16 gcn_out
  float*  assign = (float*) alloc((size_t)N * K * 4);
  float*  hat    = (float*) alloc((size_t)N * K * 4);
  float*  smallb = (float*) alloc((size_t)N * K * 4);    // gcn_out @ Ws / Wt
  float*  ASm    = (float*) alloc((size_t)N * K * 4);
  float*  deg    = (float*) alloc((size_t)N * 4);
  int*    ptr    = (int*)   alloc((size_t)(N + 1) * 4);
  int*    counts = (int*)   alloc((size_t)N * 4);
  int*    fillc  = (int*)   alloc((size_t)N * 4);
  int*    sdst   = (int*)   alloc((size_t)E * 4);
  float*  sw     = (float*) alloc((size_t)E * 4);
  __bf16* W1T    = (__bf16*)alloc((size_t)F * H * 2);    // W1 transposed bf16
  __bf16* WsT    = (__bf16*)alloc((size_t)H * K * 2);
  __bf16* WtT    = (__bf16*)alloc((size_t)H * K * 2);
  float*  acc    = (float*) alloc(64 * 4);

  hipMemsetAsync(counts, 0, (size_t)N * 4, stream);
  hipMemsetAsync(fillc,  0, (size_t)N * 4, stream);
  hipMemsetAsync(deg,    0, (size_t)N * 4, stream);
  hipMemsetAsync(acc,    0, 64 * 4, stream);

  const int tpb = 256;
  const int eblk = (E + tpb - 1) / tpb;
  const int nblk = (N + tpb - 1) / tpb;
  const int mtiles = (N + 15) / 16;
  const long long nNF = (long long)N * F;

  // weight prep (tiny, L2-resident)
  transpose_cvt_bf16<<<(F * H + tpb - 1) / tpb, tpb, 0, stream>>>(W1, W1T, F, H);
  transpose_cvt_bf16<<<(H * K + tpb - 1) / tpb, tpb, 0, stream>>>(Wst, WsT, H, K);
  transpose_cvt_bf16<<<(H * K + tpb - 1) / tpb, tpb, 0, stream>>>(Wt, WtT, H, K);

  // CSR by src (+ degrees by dst)
  edge_count<<<eblk, tpb, 0, stream>>>(esrc, edst, counts, deg, E);
  exclusive_scan<<<1, 1024, 0, stream>>>(counts, ptr, N);
  edge_fill<<<eblk, tpb, 0, stream>>>(esrc, edst, ew, ptr, fillc, sdst, sw, E);

  // ---- student branch ----
  cvt_bf16<<<(int)((nNF + tpb - 1) / tpb), tpb, 0, stream>>>(feat, XB, nNF);
  // H=256 cols: 4 waves/block x 4 tiles/wave x 16 = 256 (exact, no guards)
  gemm_wmma_bf16t<4><<<dim3(mtiles, 1), 128, 0, stream>>>(XB, W1T, XW, N, H);
  spmm_bias_selu_bf16<<<N, H, 0, stream>>>(XW, ptr, sdst, sw, b1, GB, H);
  // K=16 cols: 1 wave/block, 1 tile/wave (exact)
  gemm_wmma_bf16t<1><<<dim3(mtiles, 1), 32, 0, stream>>>(GB, WsT, smallb, N, K);
  spmm16_post<<<nblk, tpb, 0, stream>>>(smallb, ptr, sdst, sw, bs, assign, N, 1);

  // ---- teacher branch (reuses XB/GB/XW/smallb) ----
  cvt_bf16<<<(int)((nNF + tpb - 1) / tpb), tpb, 0, stream>>>(aug, XB, nNF);
  gemm_wmma_bf16t<4><<<dim3(mtiles, 1), 128, 0, stream>>>(XB, W1T, XW, N, H);
  spmm_bias_selu_bf16<<<N, H, 0, stream>>>(XW, ptr, sdst, sw, b1, GB, H);
  gemm_wmma_bf16t<1><<<dim3(mtiles, 1), 32, 0, stream>>>(GB, WtT, smallb, N, K);
  spmm16_post<<<nblk, tpb, 0, stream>>>(smallb, ptr, sdst, sw, bt, hat, N, 1);

  // AS = A @ S  (binary adjacency)
  spmm16_post<<<nblk, tpb, 0, stream>>>(assign, ptr, sdst, sw, nullptr, ASm, N, 0);

  // scalar losses
  reduce_stats<<<nblk, tpb, 0, stream>>>(assign, ASm, hat, deg, acc, N);
  finalize_loss<<<1, 1, 0, stream>>>(acc, (float*)d_out, (float)E, (float)N, (float)K);
}
